// RationaleSelectorModel_55198919688417
// MI455X (gfx1250) — matrix operand order repaired and checked by reading, not compiled, not run on hardware
//
#include <hip/hip_runtime.h>
#include <hip/hip_bf16.h>
#include <math.h>

// ---------------------------------------------------------------------------
// RationaleSelectorModel forward for MI455X (gfx1250), wave32 + WMMA.
// ---------------------------------------------------------------------------

typedef __attribute__((ext_vector_type(16))) _Float16 v16h;
typedef __attribute__((ext_vector_type(8)))  _Float16 v8h;
typedef __attribute__((ext_vector_type(8)))  float    v8f;

#define B_    8
#define T_    2048
#define D_    1024
#define H_    1365
#define NPAD_ 1376          // H padded to multiple of 16
#define NT_   (NPAD_ / 16)  // 86 N tiles
#define KT_   (D_ / 32)     // 32 K steps of 32
#define NRHO_ 5
#define NS_   5
#define NEGV  (-1e9f)
#define TAUV  1.0f

// ---------------- helpers ----------------

__device__ __forceinline__ float blockReduceSum(float v, float* sbuf) {
  // wave32 intra-wave reduce
  #pragma unroll
  for (int off = 16; off > 0; off >>= 1) v += __shfl_xor(v, off, 32);
  int lane = threadIdx.x & 31, w = threadIdx.x >> 5;
  int nw = blockDim.x >> 5;
  if (lane == 0) sbuf[w] = v;
  __syncthreads();
  v = (threadIdx.x < (unsigned)nw) ? sbuf[threadIdx.x] : 0.0f;
  if (w == 0) {
    #pragma unroll
    for (int off = 16; off > 0; off >>= 1) v += __shfl_xor(v, off, 32);
    if (lane == 0) sbuf[0] = v;
  }
  __syncthreads();
  float r = sbuf[0];
  __syncthreads();
  return r;
}

// descending bitonic sort of a[0..n) in LDS, n power-of-two, blockDim threads
__device__ __forceinline__ void bitonicSortDesc(float* a, int n) {
  for (int k = 2; k <= n; k <<= 1) {
    for (int j = k >> 1; j > 0; j >>= 1) {
      for (int i = threadIdx.x; i < n; i += blockDim.x) {
        int ix = i ^ j;
        if (ix > i) {
          float x = a[i], y = a[ix];
          bool up = ((i & k) == 0);           // segment direction
          bool sw = up ? (x < y) : (x > y);   // overall descending
          if (sw) { a[i] = y; a[ix] = x; }
        }
      }
      __syncthreads();
    }
  }
}

__device__ __forceinline__ unsigned mix32(unsigned x) {
  x ^= x >> 16; x *= 0x7feb352dU;
  x ^= x >> 15; x *= 0x846ca68bU;
  x ^= x >> 16; return x;
}
__device__ __forceinline__ float u01(unsigned x) {
  return (float)(mix32(x) >> 8) * (1.0f / 16777216.0f) + (1.0f / 33554432.0f);
}

__device__ __forceinline__ float gelu_exact(float x) {
  return 0.5f * x * (1.0f + erff(x * 0.70710678118654752f));
}

// ---------------- K1: attn-mask * LayerNorm -> f16 ----------------

__global__ __launch_bounds__(256) void k1_layernorm_f16(
    const float* __restrict__ emb, const float* __restrict__ attn,
    const float* __restrict__ lng, const float* __restrict__ lnb,
    _Float16* __restrict__ Ah) {
  __shared__ float red[32];
  int row = blockIdx.x;                 // b*T + t
  float a = attn[row];
  const float* x = emb + (size_t)row * D_;
  float ls = 0.f, lq = 0.f;
  for (int d = threadIdx.x; d < D_; d += 256) {
    float v = x[d] * a;
    ls += v; lq += v * v;
  }
  float s  = blockReduceSum(ls, red);
  float sq = blockReduceSum(lq, red);
  float mean = s * (1.0f / D_);
  float var  = sq * (1.0f / D_) - mean * mean;
  float rstd = rsqrtf(var + 1e-5f);
  _Float16* o = Ah + (size_t)row * D_;
  for (int d = threadIdx.x; d < D_; d += 256) {
    float v = (x[d] * a - mean) * rstd * lng[d] + lnb[d];
    o[d] = (_Float16)v;
  }
}

// ---------------- K2: swizzle W1 -> WMMA B-fragment layout ----------------
// B frag (16-bit, 32x16): lane (g=l>>4, c=l&15) holds column n=nt*16+c,
// halves h=0..15 <-> K = kt*32 + g*16 + h.  Stored fragment-contiguous:
// Wswz[((nt*KT + kt)*32 + lane)*16 + h]

__global__ __launch_bounds__(256) void k2_swizzle_w1(
    const float* __restrict__ W1, _Float16* __restrict__ Wswz) {
  int idx = blockIdx.x * 256 + threadIdx.x;
  int total = NT_ * KT_ * 512;
  if (idx >= total) return;
  int h    = idx & 15;
  int lane = (idx >> 4) & 31;
  int rest = idx >> 9;
  int kt   = rest % KT_;
  int nt   = rest / KT_;
  int gi = lane >> 4, c = lane & 15;
  int k = kt * 32 + gi * 16 + h;
  int n = nt * 16 + c;
  float v = (n < H_) ? W1[(size_t)k * H_ + n] : 0.0f;
  Wswz[idx] = (_Float16)v;
}

// ---------------- K3: WMMA GEMM + GELU + W2-dot -> scores ----------------

__global__ __launch_bounds__(128) void k3_scorer_wmma(
    const _Float16* __restrict__ Ah, const _Float16* __restrict__ Wswz,
    const float* __restrict__ b1, const float* __restrict__ W2,
    const float* __restrict__ b2, const float* __restrict__ attn,
    float* __restrict__ scores) {
  __shared__ __align__(16) _Float16 sA[16 * D_];   // 32 KB A tile
  __shared__ float sScore[16];

  int tid  = threadIdx.x;
  int wave = tid >> 5;
  int lane = tid & 31;
  int gi   = lane >> 4;    // K half-group
  int c    = lane & 15;    // column within tile / row for A

  if (tid < 16) sScore[tid] = 0.0f;

  // stage A tile (16 rows x 1024 halves, contiguous) via 128-bit loads
  {
    const uint4* src = (const uint4*)(Ah + (size_t)blockIdx.x * 16 * D_);
    uint4* dst = (uint4*)sA;
    for (int i = tid; i < (16 * D_) / 8; i += 128) dst[i] = src[i];
  }
  __syncthreads();

  const float b2v = b2[0];

  for (int nt = wave; nt < NT_; nt += 4) {
    v8f acc = {0.f, 0.f, 0.f, 0.f, 0.f, 0.f, 0.f, 0.f};
    const _Float16* bp = Wswz + (size_t)nt * KT_ * 512;

    for (int kt = 0; kt < KT_; ++kt) {
      // A fragment from LDS: lane row M=c; K = kt*32 + gi*16 + [0..16)
      const _Float16* arow = sA + c * D_ + kt * 32 + gi * 16;
      v8h alo = *(const v8h*)(arow);
      v8h ahi = *(const v8h*)(arow + 8);
      v16h a = __builtin_shufflevector(alo, ahi,
                                       0, 1, 2, 3, 4, 5, 6, 7,
                                       8, 9, 10, 11, 12, 13, 14, 15);
      // B fragment: contiguous 32B per lane (swizzled layout)
      v16h bfrag = *(const v16h*)(bp + kt * 512 + lane * 16);
      if (kt + 1 < KT_)
        __builtin_prefetch(bp + (kt + 1) * 512 + lane * 16, 0, 3);

      acc = __builtin_amdgcn_wmma_f32_16x16x32_f16(
          false, a, false, bfrag, (short)0, acc, false, false);
    }

    // epilogue: bias + exact GELU + W2 dot, reduce over 16 columns
    int n = nt * 16 + c;
    bool valid = (n < H_);
    float w2c = valid ? W2[n] : 0.0f;
    float b1c = valid ? b1[n] : 0.0f;
    #pragma unroll
    for (int r = 0; r < 8; ++r) {
      float x = acc[r] + b1c;
      float v = valid ? gelu_exact(x) * w2c : 0.0f;
      #pragma unroll
      for (int off = 1; off < 16; off <<= 1) v += __shfl_xor(v, off, 32);
      if (c == 0) atomicAdd(&sScore[r + gi * 8], v);   // ds_add_f32
    }
  }
  __syncthreads();

  if (tid < 16) {
    int row = blockIdx.x * 16 + tid;
    float a = attn[row];
    float s = sScore[tid] + b2v;
    scores[row] = (a == 0.0f) ? NEGV : s;
  }
}

// ---------------- K4: entmax1.5 (sort-based) -> g_soft, T_eff ----------------

__global__ __launch_bounds__(256) void k4_entmax(
    const float* __restrict__ scores, const float* __restrict__ attn,
    float* __restrict__ g_soft, float* __restrict__ T_eff) {
  __shared__ float z[T_], zs[T_], tu[T_];
  __shared__ float p1[257], p2[257];
  __shared__ float red[32];
  __shared__ int scnt;

  int b = blockIdx.x, tid = threadIdx.x;
  if (tid == 0) scnt = 0;
  const float* sc = scores + (size_t)b * T_;
  const float* at = attn + (size_t)b * T_;

  for (int t = tid; t < T_; t += 256) {
    float v = sc[t] * (0.5f / TAUV);   // z = (scores/tau)/2
    z[t] = v; zs[t] = v;
  }
  __syncthreads();
  bitonicSortDesc(zs, T_);

  // blocked cumulative sums of zs and zs^2 (8 contiguous per thread)
  int base = tid * 8;
  float s1 = 0.f, s2 = 0.f;
  #pragma unroll
  for (int i = 0; i < 8; ++i) { float v = zs[base + i]; s1 += v; s2 += v * v; }
  p1[tid + 1] = s1; p2[tid + 1] = s2;
  __syncthreads();
  if (tid == 0) {
    p1[0] = 0.f; p2[0] = 0.f;
    for (int i = 1; i <= 256; ++i) { p1[i] += p1[i - 1]; p2[i] += p2[i - 1]; }
  }
  __syncthreads();

  float c1 = p1[tid], c2 = p2[tid];
  int cnt = 0;
  #pragma unroll
  for (int i = 0; i < 8; ++i) {
    int idx = base + i;
    float v = zs[idx];
    c1 += v; c2 += v * v;
    float kk = (float)(idx + 1);
    float mean = c1 / kk, ms = c2 / kk;
    float ss = kk * (ms - mean * mean);
    float delta = fmaxf((1.0f - ss) / kk, 0.0f);
    float tau = mean - sqrtf(delta);
    tu[idx] = tau;
    cnt += (tau <= v) ? 1 : 0;
  }
  atomicAdd(&scnt, cnt);
  __syncthreads();

  int support = scnt;
  float taustar = tu[support - 1];
  __syncthreads();  // tu about to be reused

  float gls = 0.f, als = 0.f;
  for (int t = tid; t < T_; t += 256) {
    float a = at[t];
    float d = fmaxf(z[t] - taustar, 0.0f);
    float gv = d * d * a;
    tu[t] = gv;
    gls += gv; als += a;
  }
  float gsum = blockReduceSum(gls, red);
  float asum = blockReduceSum(als, red);
  float inv = 1.0f / fmaxf(gsum, 1e-6f);
  for (int t = tid; t < T_; t += 256) g_soft[(size_t)b * T_ + t] = tu[t] * inv;
  if (tid == 0) T_eff[b] = asum;
}

// ---------------- K0: zero f32 buffer ----------------

__global__ void k0_zero(float* p, int n) {
  int i = blockIdx.x * blockDim.x + threadIdx.x;
  if (i < n) p[i] = 0.0f;
}

// ---------------- K5: stochastic top-k sampling ----------------

__global__ __launch_bounds__(256) void k5_stochastic_topk(
    const float* __restrict__ scores, const float* __restrict__ attn,
    const float* __restrict__ T_eff, float* __restrict__ gh) {
  __shared__ float pert[T_], srt[T_];
  int bx = blockIdx.x;
  int j = bx / (NS_ * B_);
  int s = (bx / B_) % NS_;
  int b = bx % B_;
  int tid = threadIdx.x;

  const float* sc = scores + (size_t)b * T_;
  const float* at = attn + (size_t)b * T_;

  for (int t = tid; t < T_; t += 256) {
    unsigned seed = (unsigned)(j * 1664525u + s * 22695477u) ^
                    (unsigned)(b * 2654435761u) ^
                    (unsigned)(t * 0x9E3779B9u) ^ 0xA511E9B3u;
    float u = u01(seed);
    float noise = -logf(-logf(u + 1e-6f) + 1e-6f);
    float p = (at[t] == 0.0f) ? NEGV : sc[t] + noise * TAUV;
    pert[t] = p; srt[t] = p;
  }
  __syncthreads();
  bitonicSortDesc(srt, T_);

  float rho = 0.1f + 0.1f * (float)j;
  int k = (int)(rho * T_eff[b]);
  if (k < 1) k = 1;
  if (k > T_) k = T_;
  float thr = srt[k - 1];

  float* g = gh + ((size_t)j * B_ + b) * T_;
  for (int t = tid; t < T_; t += 256)
    if (pert[t] >= thr) atomicAdd(&g[t], 1.0f / (float)NS_);
}

// ---------------- K5b: straight-through g, rho_eff, sum_w ----------------

__global__ __launch_bounds__(256) void k5b_finalize_g(
    const float* __restrict__ gh, const float* __restrict__ g_soft,
    const float* __restrict__ attn, const float* __restrict__ T_eff,
    float* __restrict__ out_g, float* __restrict__ out_gsweep,
    float* __restrict__ out_rhoeff, float* __restrict__ sum_w) {
  __shared__ float red[32];
  int j = blockIdx.x / B_;
  int b = blockIdx.x % B_;
  int tid = threadIdx.x;
  const float* ghp = gh + ((size_t)j * B_ + b) * T_;
  const float* gsp = g_soft + (size_t)b * T_;
  const float* at  = attn + (size_t)b * T_;
  float* go = out_gsweep + ((size_t)j * B_ + b) * T_;

  float sg = 0.f, sw = 0.f;
  for (int t = tid; t < T_; t += 256) {
    float gs = gsp[t];
    float gg = (ghp[t] - gs) + gs;   // straight-through forward value
    go[t] = gg;
    if (j == NRHO_ - 1) out_g[(size_t)b * T_ + t] = gg;
    sg += gg;
    sw += at[t] * gg;
  }
  float tsg = blockReduceSum(sg, red);
  float tsw = blockReduceSum(sw, red);
  if (tid == 0) {
    out_rhoeff[j * B_ + b] = tsg / T_eff[b];
    sum_w[j * B_ + b] = tsw;
  }
}

// ---------------- K6: gather pooling (full + 5 pred reps) ----------------

__global__ __launch_bounds__(256) void k6_pool(
    const int* __restrict__ ids, const float* __restrict__ emb_table,
    const float* __restrict__ attn, const float* __restrict__ gsweep,
    const float* __restrict__ T_eff, const float* __restrict__ sum_w,
    float* __restrict__ full_rep, float* __restrict__ pred) {
  __shared__ float sw_[NRHO_ * T_];   // 40 KB
  __shared__ float sa_[T_];           //  8 KB
  int b = blockIdx.y;
  int d = blockIdx.x * 256 + threadIdx.x;
  int tid = threadIdx.x;

  for (int t = tid; t < T_; t += 256) {
    float a = attn[(size_t)b * T_ + t];
    sa_[t] = a;
    #pragma unroll
    for (int j = 0; j < NRHO_; ++j)
      sw_[j * T_ + t] = a * gsweep[((size_t)j * B_ + b) * T_ + t];
  }
  __syncthreads();

  const int* idrow = ids + (size_t)b * T_;
  float accF = 0.f;
  float acc[NRHO_] = {0.f, 0.f, 0.f, 0.f, 0.f};
  for (int t = 0; t < T_; ++t) {
    float val = emb_table[(size_t)idrow[t] * D_ + d];
    accF += val * sa_[t];
    #pragma unroll
    for (int j = 0; j < NRHO_; ++j) acc[j] += val * sw_[j * T_ + t];
  }
  full_rep[(size_t)b * D_ + d] = accF / fmaxf(T_eff[b], 1e-6f);
  #pragma unroll
  for (int j = 0; j < NRHO_; ++j)
    pred[((size_t)j * B_ + b) * D_ + d] =
        acc[j] / fmaxf(sum_w[j * B_ + b], 1e-6f);
}

// ---------------- K7: per-rho MSE ----------------

__global__ __launch_bounds__(256) void k7_loss(
    const float* __restrict__ full_rep, const float* __restrict__ pred,
    float* __restrict__ loss_out) {
  __shared__ float red[32];
  int j = blockIdx.x;
  float l = 0.f;
  for (int i = threadIdx.x; i < B_ * D_; i += 256) {
    float df = pred[(size_t)j * B_ * D_ + i] - full_rep[i];
    l += df * df;
  }
  float s = blockReduceSum(l, red);
  if (threadIdx.x == 0) loss_out[j] = s / (float)(B_ * D_);
}

__global__ void k8_avg(const float* __restrict__ loss, float* __restrict__ avg) {
  if (threadIdx.x == 0) {
    float s = 0.f;
    for (int j = 0; j < NRHO_; ++j) s += loss[j];
    *avg = s / (float)NRHO_;
  }
}

// ---------------- launch ----------------

extern "C" void kernel_launch(void* const* d_in, const int* in_sizes, int n_in,
                              void* d_out, int out_size, void* d_ws, size_t ws_size,
                              hipStream_t stream) {
  (void)in_sizes; (void)n_in; (void)out_size; (void)ws_size;

  const int*   ids  = (const int*)d_in[0];
  const float* emb  = (const float*)d_in[1];
  const float* attn = (const float*)d_in[2];
  const float* lng  = (const float*)d_in[3];
  const float* lnb  = (const float*)d_in[4];
  const float* W1   = (const float*)d_in[5];
  const float* b1   = (const float*)d_in[6];
  const float* W2   = (const float*)d_in[7];
  const float* b2   = (const float*)d_in[8];
  const float* embt = (const float*)d_in[9];

  float* out = (float*)d_out;
  float* out_g      = out;                         // [B,T]
  float* out_gsweep = out + B_ * T_;               // [5,B,T]
  float* out_recon  = out + B_ * T_ * (1 + NRHO_); // [1]
  float* out_loss   = out_recon + 1;               // [5]
  float* out_rhoeff = out_loss + NRHO_;            // [5,B]

  // workspace carve-up (256B aligned)
  char* ws = (char*)d_ws;
  size_t off = 0;
  auto alloc = [&](size_t bytes) {
    char* p = ws + off;
    off += (bytes + 255) & ~(size_t)255;
    return p;
  };
  _Float16* Ah     = (_Float16*)alloc((size_t)B_ * T_ * D_ * 2);      // 32 MB
  _Float16* Wswz   = (_Float16*)alloc((size_t)NT_ * KT_ * 512 * 2);   // 2.75 MB
  float*    scores = (float*)alloc((size_t)B_ * T_ * 4);
  float*    g_soft = (float*)alloc((size_t)B_ * T_ * 4);
  float*    T_eff  = (float*)alloc(B_ * 4);
  float*    gh     = (float*)alloc((size_t)NRHO_ * B_ * T_ * 4);
  float*    sum_w  = (float*)alloc(NRHO_ * B_ * 4);
  float*    frep   = (float*)alloc((size_t)B_ * D_ * 4);
  float*    pred   = (float*)alloc((size_t)NRHO_ * B_ * D_ * 4);

  // K1: LayerNorm -> f16
  k1_layernorm_f16<<<B_ * T_, 256, 0, stream>>>(emb, attn, lng, lnb, Ah);
  // K2: swizzle W1
  {
    int total = NT_ * KT_ * 512;
    k2_swizzle_w1<<<(total + 255) / 256, 256, 0, stream>>>(W1, Wswz);
  }
  // K3: WMMA scorer
  k3_scorer_wmma<<<(B_ * T_) / 16, 128, 0, stream>>>(
      Ah, Wswz, b1, W2, b2, attn, scores);
  // K4: entmax -> g_soft, T_eff
  k4_entmax<<<B_, 256, 0, stream>>>(scores, attn, g_soft, T_eff);
  // K0: zero g_hard accumulators
  {
    int n = NRHO_ * B_ * T_;
    k0_zero<<<(n + 255) / 256, 256, 0, stream>>>(gh, n);
  }
  // K5: stochastic top-k (rho x sample x batch blocks)
  k5_stochastic_topk<<<NRHO_ * NS_ * B_, 256, 0, stream>>>(
      scores, attn, T_eff, gh);
  // K5b: straight-through g + rho_eff + sum_w
  k5b_finalize_g<<<NRHO_ * B_, 256, 0, stream>>>(
      gh, g_soft, attn, T_eff, out_g, out_gsweep, out_rhoeff, sum_w);
  // K6: pooling (full + pred reps)
  {
    dim3 grid(D_ / 256, B_);
    k6_pool<<<grid, 256, 0, stream>>>(
        ids, embt, attn, out_gsweep, T_eff, sum_w, frep, pred);
  }
  // K7/K8: losses
  k7_loss<<<NRHO_, 256, 0, stream>>>(frep, pred, out_loss);
  k8_avg<<<1, 32, 0, stream>>>(out_loss, out_recon);
}
